// Quantizer_27573690040472
// MI455X (gfx1250) — compile-verified, matrix operand order checked by reading, not run on hardware
//
#include <hip/hip_runtime.h>
#include <hip/hip_bf16.h>
#include <cstdint>
#include <cstddef>

typedef _Float16 half8 __attribute__((ext_vector_type(8)));
typedef _Float16 v16h  __attribute__((ext_vector_type(16)));
typedef float    v8f   __attribute__((ext_vector_type(8)));

#define B_ 8
#define T_ 4096
#define H_ 1024
#define Q_ 8
#define K_ 1024
#define D_ 128
#define NT_ (B_*T_)          /* 32768 tokens */
#define RMS_EPS_ 1e-6f

// ---------------- kernel 0a: half squared norms of codewords ----------------
__global__ __launch_bounds__(256)
void k_c2(const float* __restrict__ cb, float* __restrict__ hc2) {
    int i = blockIdx.x * blockDim.x + threadIdx.x;   // q*K + k
    if (i >= Q_ * K_) return;
    const float* r = cb + (size_t)i * D_;
    float s = 0.f;
    #pragma unroll 8
    for (int d = 0; d < D_; ++d) { float v = r[d]; s += v * v; }
    hc2[i] = 0.5f * s;
}

// ---------------- kernel 0b: split codebooks into f16 hi + lo ----------------
__global__ __launch_bounds__(256)
void k_split(const float* __restrict__ cb,
             _Float16* __restrict__ hi, _Float16* __restrict__ lo) {
    int i = (blockIdx.x * blockDim.x + threadIdx.x) * 4;
    const float4 v = *(const float4*)(cb + i);
    float vv[4] = {v.x, v.y, v.z, v.w};
    #pragma unroll
    for (int j = 0; j < 4; ++j) {
        _Float16 h = (_Float16)vv[j];
        hi[i + j] = h;
        lo[i + j] = (_Float16)(vv[j] - (float)h);
    }
}

// ---------------- kernel 1: WMMA nearest-codeword search ----------------
// grid = (NT/16, Q), block = 256 (8 wave32). Each block: 16 tokens x 1 q-group.
// score(m,n) = xc(m,n) - 0.5*c2(n); dist argmin == score argmax.
__global__ __launch_bounds__(256)
void k_argmin(const float* __restrict__ x,
              const _Float16* __restrict__ cb_hi,
              const _Float16* __restrict__ cb_lo,
              const float* __restrict__ hc2,
              int* __restrict__ idx_out) {
    __shared__ _Float16 xhi[16][D_];
    __shared__ _Float16 xlo[16][D_];
    __shared__ float red_s[8][16][16];
    __shared__ int   red_i[8][16][16];

    const int tid  = threadIdx.x;
    const int lane = tid & 31;
    const int w    = tid >> 5;          // wave id 0..7
    const int q    = blockIdx.y;
    const int bt0  = blockIdx.x * 16;   // first token of tile

    // Stage the 16x128 x-tile for this q into LDS as f16 hi/lo.
    {
        const int tok = tid >> 4;            // 0..15
        const int d0  = (tid & 15) * 8;      // 0..120
        const float* p = x + (size_t)(bt0 + tok) * H_ + q * D_ + d0;
        float4 a = *(const float4*)(p);
        float4 b = *(const float4*)(p + 4);
        float vv[8] = {a.x, a.y, a.z, a.w, b.x, b.y, b.z, b.w};
        #pragma unroll
        for (int j = 0; j < 8; ++j) {
            _Float16 h = (_Float16)vv[j];
            xhi[tok][d0 + j] = h;
            xlo[tok][d0 + j] = (_Float16)(vv[j] - (float)h);
        }
    }
    __syncthreads();

    // A fragments in the documented 16-bit A layout:
    //  lanes 0-15 : M=lane,    K = kb*32 + {0..7, 16..23}
    //  lanes 16-31: M=lane-16, K = kb*32 + {8..15, 24..31}
    const int m    = lane & 15;
    const int koff = (lane < 16) ? 0 : 8;
    v16h a_hi[4], a_lo[4];
    #pragma unroll
    for (int kb = 0; kb < 4; ++kb) {
        const int k0 = kb * 32 + koff;
        half8 h0 = *(const half8*)&xhi[m][k0];
        half8 h1 = *(const half8*)&xhi[m][k0 + 16];
        a_hi[kb] = __builtin_shufflevector(h0, h1, 0,1,2,3,4,5,6,7,8,9,10,11,12,13,14,15);
        half8 l0 = *(const half8*)&xlo[m][k0];
        half8 l1 = *(const half8*)&xlo[m][k0 + 16];
        a_lo[kb] = __builtin_shufflevector(l0, l1, 0,1,2,3,4,5,6,7,8,9,10,11,12,13,14,15);
    }

    float bestv[8];
    int   besti[8];
    #pragma unroll
    for (int r = 0; r < 8; ++r) { bestv[r] = -3.4e38f; besti[r] = 0; }

    // Each wave covers codewords [w*128, w*128+128) in 8 tiles of 16.
    for (int it = 0; it < 8; ++it) {
        const int nbase = (w * 8 + it) * 16;
        const int n = nbase + (lane & 15);        // this lane's column (C layout: N = lane%16)
        const float c2h = hc2[q * K_ + n];
        v8f acc;
        #pragma unroll
        for (int r = 0; r < 8; ++r) acc[r] = -c2h;   // fold -0.5*c2 into C

        const _Float16* bh = cb_hi + (size_t)(q * K_ + n) * D_;
        const _Float16* bl = cb_lo + (size_t)(q * K_ + n) * D_;
        #pragma unroll
        for (int kb = 0; kb < 4; ++kb) {
            const int k0 = kb * 32 + koff;
            half8 c0 = *(const half8*)(bh + k0);
            half8 c1 = *(const half8*)(bh + k0 + 16);
            v16h b_hi = __builtin_shufflevector(c0, c1, 0,1,2,3,4,5,6,7,8,9,10,11,12,13,14,15);
            half8 e0 = *(const half8*)(bl + k0);
            half8 e1 = *(const half8*)(bl + k0 + 16);
            v16h b_lo = __builtin_shufflevector(e0, e1, 0,1,2,3,4,5,6,7,8,9,10,11,12,13,14,15);
            // 3-term error-compensated f16 product: hi*hi + hi*lo + lo*hi
            acc = __builtin_amdgcn_wmma_f32_16x16x32_f16(false, a_hi[kb], false, b_hi, (short)0, acc, false, false);
            acc = __builtin_amdgcn_wmma_f32_16x16x32_f16(false, a_hi[kb], false, b_lo, (short)0, acc, false, false);
            acc = __builtin_amdgcn_wmma_f32_16x16x32_f16(false, a_lo[kb], false, b_hi, (short)0, acc, false, false);
        }
        // C layout: VGPR r, lanes 0-15 -> (M=r, N=lane); lanes 16-31 -> (M=r+8, N=lane-16)
        #pragma unroll
        for (int r = 0; r < 8; ++r) {
            if (acc[r] > bestv[r]) { bestv[r] = acc[r]; besti[r] = n; }
        }
    }

    // Per-wave candidates -> LDS
    #pragma unroll
    for (int r = 0; r < 8; ++r) {
        const int tok = r + ((lane < 16) ? 0 : 8);
        red_s[w][tok][lane & 15] = bestv[r];
        red_i[w][tok][lane & 15] = besti[r];
    }
    __syncthreads();

    // Final reduction: one thread per token scans 8 waves x 16 lanes.
    if (tid < 16) {
        float bs = -3.4e38f; int bi = 0x7fffffff;
        for (int ww = 0; ww < 8; ++ww)
            for (int c = 0; c < 16; ++c) {
                float s = red_s[ww][tid][c];
                int   i = red_i[ww][tid][c];
                if (s > bs || (s == bs && i < bi)) { bs = s; bi = i; }
            }
        idx_out[(size_t)(bt0 + tid) * Q_ + q] = bi;
    }
}

// ---------------- kernel 2: gather winning codewords + RMSNorm ----------------
// grid = NT, block = 256. 32 threads per q-group, 4 dims each (float4).
__global__ __launch_bounds__(256)
void k_gather_norm(const float* __restrict__ cb,
                   const int* __restrict__ idx,
                   const float* __restrict__ wgt,
                   float* __restrict__ out) {
    __shared__ float red[256];
    const int token = blockIdx.x;
    const int tid   = threadIdx.x;
    const int q     = tid >> 5;
    const int dd    = (tid & 31) * 4;
    const int id    = idx[(size_t)token * Q_ + q];
    const float4 v  = *(const float4*)(cb + (size_t)(q * K_ + id) * D_ + dd);
    red[tid] = v.x*v.x + v.y*v.y + v.z*v.z + v.w*v.w;
    __syncthreads();
    #pragma unroll
    for (int s = 128; s > 0; s >>= 1) {
        if (tid < s) red[tid] += red[tid + s];
        __syncthreads();
    }
    const float scale = rsqrtf(red[0] * (1.0f / H_) + RMS_EPS_);
    const int h = q * D_ + dd;
    float4 o;
    o.x = v.x * scale * wgt[h];
    o.y = v.y * scale * wgt[h + 1];
    o.z = v.z * scale * wgt[h + 2];
    o.w = v.w * scale * wgt[h + 3];
    *(float4*)(out + (size_t)token * H_ + h) = o;
}

// ---------------- launcher ----------------
extern "C" void kernel_launch(void* const* d_in, const int* in_sizes, int n_in,
                              void* d_out, int out_size, void* d_ws, size_t ws_size,
                              hipStream_t stream) {
    (void)in_sizes; (void)n_in; (void)out_size; (void)ws_size;
    const float* x   = (const float*)d_in[0];
    const float* cb  = (const float*)d_in[1];
    const float* wgt = (const float*)d_in[2];

    float* out     = (float*)d_out;
    int*   idx_out = (int*)(out + (size_t)NT_ * H_);   // idx region follows normed

    // workspace layout: [hc2: 32KB][cb_hi: 2MB][cb_lo: 2MB]
    char* ws = (char*)d_ws;
    float*    hc2   = (float*)ws;
    _Float16* cb_hi = (_Float16*)(ws + 32768);
    _Float16* cb_lo = (_Float16*)(ws + 32768 + (size_t)Q_ * K_ * D_ * sizeof(_Float16));

    k_c2<<<(Q_ * K_ + 255) / 256, 256, 0, stream>>>(cb, hc2);
    k_split<<<(Q_ * K_ * D_ / 4) / 256, 256, 0, stream>>>(cb, cb_hi, cb_lo);

    dim3 g1(NT_ / 16, Q_);
    k_argmin<<<g1, 256, 0, stream>>>(x, cb_hi, cb_lo, hc2, idx_out);

    k_gather_norm<<<NT_, 256, 0, stream>>>(cb, idx_out, wgt, out);
}
